// HROMAttention_13451837571088
// MI455X (gfx1250) — matrix operand order repaired and checked by reading, not compiled, not run on hardware
//
#include <hip/hip_runtime.h>
#include <hip/hip_bf16.h>
#include <math.h>

// ---------------------------------------------------------------------------
// Problem constants (from reference): B=4, T=2048, DIM=512, H=8, HD=64
// ---------------------------------------------------------------------------
constexpr int kDim   = 512;
constexpr int kHeads = 8;
constexpr int kHd    = 64;
constexpr int kB     = 4;
constexpr int kT     = 2048;
constexpr int kBT    = kB * kT;     // 8192 rows
constexpr int k3Dim  = 3 * kDim;    // 1536

typedef __attribute__((ext_vector_type(16))) _Float16 v16h;
typedef __attribute__((ext_vector_type(8)))  _Float16 v8h;
typedef __attribute__((ext_vector_type(8)))  float    v8f;

__device__ __forceinline__ v16h cat8(v8h a, v8h b) {
  return __builtin_shufflevector(a, b, 0,1,2,3,4,5,6,7,8,9,10,11,12,13,14,15);
}

// A-fragment (16x32 f16, M x K), source row-major [M][K] (K contiguous).
// ISA layout: lane r (0-15) holds K = hi*8..hi*8+7 and K = 16+hi*8..16+hi*8+7.
__device__ __forceinline__ v16h load_fragA(const _Float16* a, int ld) {
  const int lane = threadIdx.x & 31;
  const int r = lane & 15, hi = lane >> 4;
  const _Float16* p = a + (size_t)r * ld + hi * 8;
  return cat8(*(const v8h*)(p), *(const v8h*)(p + 16));
}

// B-fragment (32x16 f16, K x N), source = B^T row-major [N][K].
// ISA layout: lane n (0-15) holds K = hi*8..hi*8+7 and K = 16+hi*8..16+hi*8+7 (col n = lane&15).
__device__ __forceinline__ v16h load_fragB(const _Float16* bt, int ld) {
  const int lane = threadIdx.x & 31;
  const int r = lane & 15, hi = lane >> 4;
  const _Float16* p = bt + (size_t)r * ld + hi * 8;
  return cat8(*(const v8h*)(p), *(const v8h*)(p + 16));
}

__device__ __forceinline__ v8f wmma16(v16h a, v16h b, v8f c) {
  return __builtin_amdgcn_wmma_f32_16x16x32_f16(
      /*neg_a=*/false, a, /*neg_b=*/false, b,
      /*c_mod=*/(short)0, c, /*reuse_a=*/false, /*reuse_b=*/false);
}

// ---------------------------------------------------------------------------
// fp32 -> f16 convert
// ---------------------------------------------------------------------------
__global__ void cvt_f32_to_f16_kernel(const float* __restrict__ in,
                                      _Float16* __restrict__ out, int n) {
  int i = blockIdx.x * blockDim.x + threadIdx.x;
  if (i < n) out[i] = (_Float16)in[i];
}

// ---------------------------------------------------------------------------
// C[M][N] (fp32) = A[M][K] (f16) @ BT[N][K]^T (f16) + bias[N]
// block = 128 (4 waves); wave -> one 16-row tile; block covers 64 rows x 64 cols.
// All 5 fragments of an iteration live in distinct registers so the 10 b128
// loads clause together and s_wait_loadcnt can retire incrementally.
// ---------------------------------------------------------------------------
__global__ void __launch_bounds__(128)
wmma_gemm_bias_kernel(const _Float16* __restrict__ A, const _Float16* __restrict__ BT,
                      const float* __restrict__ bias, float* __restrict__ C,
                      int M, int N, int K) {
  const int wave = threadIdx.x >> 5;
  const int lane = threadIdx.x & 31;
  const int row0 = (blockIdx.y * 4 + wave) * 16;
  const int n0   = blockIdx.x * 64;
  const _Float16* arow = A + (size_t)row0 * K;
  const _Float16* bt0  = BT + (size_t)(n0     ) * K;
  const _Float16* bt1  = BT + (size_t)(n0 + 16) * K;
  const _Float16* bt2  = BT + (size_t)(n0 + 32) * K;
  const _Float16* bt3  = BT + (size_t)(n0 + 48) * K;

  v8f acc0 = {}, acc1 = {}, acc2 = {}, acc3 = {};
  for (int k0 = 0; k0 < K; k0 += 32) {
    const v16h a  = load_fragA(arow + k0, K);
    const v16h b0 = load_fragB(bt0 + k0, K);
    const v16h b1 = load_fragB(bt1 + k0, K);
    const v16h b2 = load_fragB(bt2 + k0, K);
    const v16h b3 = load_fragB(bt3 + k0, K);
    if (k0 + 32 < K) {
      __builtin_prefetch(arow + k0 + 32, 0, 3);                 // global_prefetch
      __builtin_prefetch(bt0 + k0 + 32, 0, 3);
      __builtin_prefetch(bt2 + k0 + 32, 0, 3);
    }
    acc0 = wmma16(a, b0, acc0);
    acc1 = wmma16(a, b1, acc1);
    acc2 = wmma16(a, b2, acc2);
    acc3 = wmma16(a, b3, acc3);
  }

  const int r = lane & 15, hi = lane >> 4;
  const v8f accs[4] = {acc0, acc1, acc2, acc3};
#pragma unroll
  for (int j = 0; j < 4; ++j) {
    const int col = n0 + 16 * j + r;
    const float bv = bias[col];
#pragma unroll
    for (int c = 0; c < 8; ++c)
      C[(size_t)(row0 + c + 8 * hi) * N + col] = accs[j][c] + bv;
  }
}

// ---------------------------------------------------------------------------
// RoPE + scatter: qkv[BT][1536] fp32 -> Qh,Kh [B,H,T,64] f16 (rotated),
//                 Vt [B,H,64,T] f16 (transposed for B-operand of PV)
// ---------------------------------------------------------------------------
__global__ void rope_scatter_kernel(const float* __restrict__ qkv,
                                    _Float16* __restrict__ Qh,
                                    _Float16* __restrict__ Kh,
                                    _Float16* __restrict__ Vt) {
  const int idx = blockIdx.x * blockDim.x + threadIdx.x;
  const int d = idx & 63;
  const int h = (idx >> 6) & 7;
  const int t = (idx >> 9) & (kT - 1);
  const int b = idx >> 20;
  const float* row = qkv + (size_t)(b * kT + t) * k3Dim;

  const int c  = h * 64 + d;
  const int j  = d & 31;
  const int d2 = (d < 32) ? d + 32 : d - 32;
  const float sgn = (d < 32) ? -1.f : 1.f;
  // inv_freq = 10000^(-2j/64); ln(10000) = 9.210340371976184
  const float inv_freq = __expf(-(float)j * (9.210340371976184f / 32.f));
  const float ang = (float)t * inv_freq;
  const float cs = cosf(ang), sn = sinf(ang);

  const float q = row[c],        qp = row[h * 64 + d2];
  const float k = row[kDim + c], kp = row[kDim + h * 64 + d2];
  const float v = row[2 * kDim + c];

  const size_t qi = ((size_t)(b * kHeads + h) * kT + t) * kHd + d;
  Qh[qi] = (_Float16)(q * cs + sgn * qp * sn);
  Kh[qi] = (_Float16)(k * cs + sgn * kp * sn);
  Vt[((size_t)(b * kHeads + h) * kHd + d) * kT + t] = (_Float16)v;
}

// ---------------------------------------------------------------------------
// Flash attention, cooperative + double-buffered.
// block = 128 (4 waves), grid = (T/64, B*H). Per 32-token chunk the block
// stages K (32x64, 4KB) and V^T (64x32, 4KB) ONCE into LDS (16 b128 global
// loads + 16 ds_store_b128), consumed by all 4 waves as ds_load_b128
// fragments. Next chunk's global loads are issued before the softmax VALU
// work and committed to the alternate LDS buffer at the end of the iteration:
// one s_barrier per chunk, loads overlap compute. Scores never leave VGPRs;
// P transposes to A-operand layout through a per-wave LDS slab (same-wave DS
// ops complete in order, so no extra barrier).
// ---------------------------------------------------------------------------
__global__ void __launch_bounds__(128)
flash_attn_kernel(const _Float16* __restrict__ Qh, const _Float16* __restrict__ Kh,
                  const _Float16* __restrict__ Vt, const float* __restrict__ mask,
                  _Float16* __restrict__ Oh) {
  __shared__ __align__(16) _Float16 kbuf[2][32 * 64];  // [token][d]
  __shared__ __align__(16) _Float16 vbuf[2][64 * 32];  // [d][token]
  __shared__ __align__(16) _Float16 pbuf[4][16 * 32];  // per-wave P slab

  const int tid  = threadIdx.x;
  const int wave = tid >> 5;
  const int lane = tid & 31;
  const int r = lane & 15, hi = lane >> 4;
  const int bh = blockIdx.y;
  const int b = bh >> 3, h = bh & 7;
  const int q0 = (blockIdx.x * 4 + wave) * 16;

  const _Float16* Qbase = Qh + ((size_t)bh * kT + q0) * kHd;
  const _Float16* Kbase = Kh + (size_t)bh * kT * kHd;
  const _Float16* Vbase = Vt + (size_t)bh * kHd * kT;
  const float* mvec = mask + b * kT;
  _Float16* pb = &pbuf[wave][0];

  const v16h qa0 = load_fragA(Qbase, kHd);       // d = 0..31
  const v16h qa1 = load_fragA(Qbase + 32, kHd);  // d = 32..63

  v8f o0 = {}, o1 = {}, o2 = {}, o3 = {};
  float m_run[8], l_run[8];
#pragma unroll
  for (int c = 0; c < 8; ++c) { m_run[c] = -1e30f; l_run[c] = 0.f; }
  const float scale = 0.125f;  // 1/sqrt(64)

  // cooperative staging: 256 uint4 per chunk for K and for V, 2 each/thread
  uint4 kr0, kr1, vr0, vr1;
  auto ld_stage = [&](int k0) {
    const uint4* ks = (const uint4*)(Kbase + (size_t)k0 * kHd);  // contiguous 4KB
    kr0 = ks[tid];
    kr1 = ks[tid + 128];
    vr0 = *((const uint4*)(Vbase + (size_t)(tid >> 2) * kT + k0) + (tid & 3));
    vr1 = *((const uint4*)(Vbase + (size_t)((tid >> 2) + 32) * kT + k0) + (tid & 3));
  };
  auto st_stage = [&](int s) {
    ((uint4*)kbuf[s])[tid]       = kr0;
    ((uint4*)kbuf[s])[tid + 128] = kr1;
    ((uint4*)vbuf[s])[tid]       = vr0;
    ((uint4*)vbuf[s])[tid + 128] = vr1;
  };

  ld_stage(0);
  st_stage(0);
  int s = 0;
  for (int k0 = 0; k0 < kT; k0 += 32, s ^= 1) {
    __syncthreads();  // buffer s staged; buffer s^1 free to overwrite
    const bool more = (k0 + 32) < kT;
    if (more) ld_stage(k0 + 32);  // issue next chunk's global loads now

    const _Float16* kcs = kbuf[s];
    const _Float16* vcs = vbuf[s];

    // scores S[16q][32k] = Q(16x64) @ K^T, two 16x16 f32 tiles, frags from LDS
    v8f s0 = {}, s1 = {};
    {
      v16h kb;
      kb = load_fragB(kcs, 64);               s0 = wmma16(qa0, kb, s0);
      kb = load_fragB(kcs + 32, 64);          s0 = wmma16(qa1, kb, s0);
      kb = load_fragB(kcs + 16 * 64, 64);     s1 = wmma16(qa0, kb, s1);
      kb = load_fragB(kcs + 16 * 64 + 32, 64);s1 = wmma16(qa1, kb, s1);
    }
    const float mk0 = mvec[k0 + r];
    const float mk1 = mvec[k0 + 16 + r];

    float corr[8];
#pragma unroll
    for (int c = 0; c < 8; ++c) {
      float v0 = s0[c] * scale + mk0;
      float v1 = s1[c] * scale + mk1;
      float mx = fmaxf(v0, v1);
#pragma unroll
      for (int off = 8; off >= 1; off >>= 1)        // 16-lane (half-wave) row max
        mx = fmaxf(mx, __shfl_xor(mx, off, 32));
      const float mnew = fmaxf(m_run[c], mx);
      corr[c] = __expf(m_run[c] - mnew);
      const float p0 = __expf(v0 - mnew);
      const float p1 = __expf(v1 - mnew);
      float ps = p0 + p1;
#pragma unroll
      for (int off = 8; off >= 1; off >>= 1)        // row sum
        ps += __shfl_xor(ps, off, 32);
      l_run[c] = l_run[c] * corr[c] + ps;
      m_run[c] = mnew;
      // stage P (f16) row-major [16][32] in this wave's LDS slab
      pb[(c + 8 * hi) * 32 + r]      = (_Float16)p0;
      pb[(c + 8 * hi) * 32 + 16 + r] = (_Float16)p1;
    }
#pragma unroll
    for (int c = 0; c < 8; ++c) {
      o0[c] *= corr[c]; o1[c] *= corr[c]; o2[c] *= corr[c]; o3[c] *= corr[c];
    }
    // same-wave LDS RAW: hardware DS ops are in order; just pin compiler order
    asm volatile("" ::: "memory");
    const v16h pa = load_fragA(pb, 32);   // P as 16x32 A operand
    asm volatile("" ::: "memory");

    // O(16x64) += P(16x32) @ V(32x64): vbuf rows are d, cols are tokens
    o0 = wmma16(pa, load_fragB(vcs,           32), o0);
    o1 = wmma16(pa, load_fragB(vcs + 16 * 32, 32), o1);
    o2 = wmma16(pa, load_fragB(vcs + 32 * 32, 32), o2);
    o3 = wmma16(pa, load_fragB(vcs + 48 * 32, 32), o3);

    if (more) st_stage(s ^ 1);  // commit next chunk (waits its loadcnt here)
  }

  // epilogue: normalize, write f16 [BT][512] for the projection GEMM
#pragma unroll
  for (int c = 0; c < 8; ++c) {
    const float inv = 1.f / l_run[c];
    const size_t rowo = (size_t)(b * kT + q0 + c + 8 * hi) * kDim + h * kHd;
    Oh[rowo + r]      = (_Float16)(o0[c] * inv);
    Oh[rowo + 16 + r] = (_Float16)(o1[c] * inv);
    Oh[rowo + 32 + r] = (_Float16)(o2[c] * inv);
    Oh[rowo + 48 + r] = (_Float16)(o3[c] * inv);
  }
}

// ---------------------------------------------------------------------------
// Launch
// ---------------------------------------------------------------------------
extern "C" void kernel_launch(void* const* d_in, const int* in_sizes, int n_in,
                              void* d_out, int out_size, void* d_ws, size_t ws_size,
                              hipStream_t stream) {
  (void)in_sizes; (void)n_in; (void)out_size; (void)ws_size;
  const float* x      = (const float*)d_in[0];
  const float* mask   = (const float*)d_in[1];
  const float* qkv_w  = (const float*)d_in[2];
  const float* qkv_b  = (const float*)d_in[3];
  const float* proj_w = (const float*)d_in[4];
  const float* proj_b = (const float*)d_in[5];
  float* out = (float*)d_out;

  char* ws = (char*)d_ws;
  auto alloc = [&](size_t bytes) {
    char* p = ws;
    ws += (bytes + 255) & ~(size_t)255;
    return p;
  };
  _Float16* xh     = (_Float16*)alloc((size_t)kBT * kDim * 2);          // x in f16
  _Float16* wqkvh  = (_Float16*)alloc((size_t)k3Dim * kDim * 2);        // qkv_w f16 (already [N][K])
  _Float16* wprojh = (_Float16*)alloc((size_t)kDim * kDim * 2);         // proj_w f16 (already [N][K])
  float*    qkvf   = (float*)   alloc((size_t)kBT * k3Dim * 4);         // fused qkv, fp32
  _Float16* Qh     = (_Float16*)alloc((size_t)kB * kHeads * kT * kHd * 2);
  _Float16* Kh     = (_Float16*)alloc((size_t)kB * kHeads * kT * kHd * 2);
  _Float16* Vt     = (_Float16*)alloc((size_t)kB * kHeads * kHd * kT * 2);
  _Float16* Oh     = (_Float16*)alloc((size_t)kBT * kDim * 2);          // attn out f16

  // 1) converts
  {
    int n = kBT * kDim;
    cvt_f32_to_f16_kernel<<<(n + 255) / 256, 256, 0, stream>>>(x, xh, n);
    n = k3Dim * kDim;
    cvt_f32_to_f16_kernel<<<(n + 255) / 256, 256, 0, stream>>>(qkv_w, wqkvh, n);
    n = kDim * kDim;
    cvt_f32_to_f16_kernel<<<(n + 255) / 256, 256, 0, stream>>>(proj_w, wprojh, n);
  }
  // 2) QKV GEMM + bias
  wmma_gemm_bias_kernel<<<dim3(k3Dim / 64, kBT / 64), 128, 0, stream>>>(
      xh, wqkvh, qkv_b, qkvf, kBT, k3Dim, kDim);
  // 3) RoPE + layout scatter
  {
    int n = kB * kT * kDim;  // one thread per (b,t,h,d)
    rope_scatter_kernel<<<n / 256, 256, 0, stream>>>(qkvf, Qh, Kh, Vt);
  }
  // 4) flash attention
  flash_attn_kernel<<<dim3(kT / 64, kB * kHeads), 128, 0, stream>>>(
      Qh, Kh, Vt, mask, Oh);
  // 5) output projection + bias -> fp32 result
  wmma_gemm_bias_kernel<<<dim3(kDim / 64, kBT / 64), 128, 0, stream>>>(
      Oh, wprojh, proj_b, out, kBT, kDim, kDim);
}
